// WindowObjectDetector_12859132084490
// MI455X (gfx1250) — matrix operand order repaired and checked by reading, not compiled, not run on hardware
//
#include <hip/hip_runtime.h>
#include <hip/hip_bf16.h>

typedef __attribute__((ext_vector_type(16))) _Float16 v16h;
typedef __attribute__((ext_vector_type(8)))  float    v8f;

#define RGRID   61
#define NCELL   3721      // 61*61 patches / grid cells
#define KCLS    10
#define DIM     12288     // 64*64*3
#define NCHUNK  384       // DIM / 32
#define NTILE   233       // ceil(3721/16)
#define NSEG    3722      // NCELL + 1 segments
#define BIGV    3723      // R*C + 2, matches reference
#define THRESH  0.7f

static __device__ __forceinline__ int imin2(int a, int b) { return a < b ? a : b; }

// ---------------------------------------------------------------------------
// Pack W (12288 x 10, f32) into zero-padded f16 WMMA B-fragments:
// Wh[ck*512 + lane*16 + e] = W[(ck*32 + 16*hi + e)*10 + lo]  (lo<10, else 0)
// where lane = lo + 16*hi.  One contiguous v16h load per lane in the GEMM.
// ---------------------------------------------------------------------------
__global__ void prep_w_kernel(const float* __restrict__ W, _Float16* __restrict__ Wh) {
    int t = blockIdx.x * blockDim.x + threadIdx.x;
    if (t >= NCHUNK * 512) return;
    int ck   = t >> 9;
    int idx  = t & 511;
    int lane = idx >> 4;
    int e    = idx & 15;
    int lo   = lane & 15;
    int hi   = lane >> 4;
    int kg   = ck * 32 + hi * 16 + e;               // B-matrix row K
    float v  = (lo < KCLS) ? W[kg * KCLS + lo] : 0.0f;
    Wh[t] = (_Float16)v;
}

// ---------------------------------------------------------------------------
// Patch-GEMM (WMMA f16->f32) + bias + softmax.
// One block = 16 patches x 16 cols (10 valid). 8 waves split the 384 K-chunks;
// each wave runs TWO independent accumulator streams (breaks the WMMA->WMMA
// RAW chain, 5-slot hazard for f16 WMMA) over v_wmma_f32_16x16x32_f16.
// Partials reduce through LDS, then softmax over the 10 classes.
// ---------------------------------------------------------------------------
__global__ __launch_bounds__(256) void gemm_softmax_kernel(
    const float* __restrict__ img, const _Float16* __restrict__ Wh,
    const float* __restrict__ bias, float* __restrict__ prob_out) {
    __shared__ float part[8 * 256];
    __shared__ float logits[256];

    const int tile = blockIdx.x;
    const int tid  = threadIdx.x;
    const int lane = tid & 31;
    const int wave = tid >> 5;
    const int lo   = lane & 15;
    const int hi   = lane >> 4;

    // A-fragment row M = lo -> patch index (clamped for the ragged last tile)
    int p = tile * 16 + lo;
    if (p > NCELL - 1) p = NCELL - 1;
    const int pr = p / RGRID;
    const int pc = p % RGRID;
    // image is (1024,1024,3) row-major: base of patch = ((pr*16)*1024 + pc*16)*3
    const int patch_base = (pr * 16) * 3072 + pc * 48;   // 3072 = 1024*3

    v8f acc0 = {0.f, 0.f, 0.f, 0.f, 0.f, 0.f, 0.f, 0.f};
    v8f acc1 = acc0;

    // wave's chunks: ck = wave + 8*t, t = 0..47; process pairs (t, t+1)
    for (int t = 0; t < NCHUNK / 8; t += 2) {
        const int ck0 = wave + 8 * t;
        const int ck1 = ck0 + 8;

        // ---- stream 0 ----
        {
            const int kbase = ck0 * 32;
            const int i   = kbase / 192;    // patch row (chunk never straddles: 32|192)
            const int rem = kbase % 192;
            const float* __restrict__ src = img + patch_base + i * 3072 + rem;
            v16h a;
#pragma unroll
            for (int e = 0; e < 16; ++e) {
                const int kic = e + 16 * hi;
                a[e] = (_Float16)src[kic];
            }
            const v16h b = *(const v16h*)(Wh + (ck0 << 9) + (lane << 4));
            acc0 = __builtin_amdgcn_wmma_f32_16x16x32_f16(
                false, a, false, b, (short)0, acc0, false, false);
        }
        // ---- stream 1 ----
        {
            const int kbase = ck1 * 32;
            const int i   = kbase / 192;
            const int rem = kbase % 192;
            const float* __restrict__ src = img + patch_base + i * 3072 + rem;
            v16h a;
#pragma unroll
            for (int e = 0; e < 16; ++e) {
                const int kic = e + 16 * hi;
                a[e] = (_Float16)src[kic];
            }
            const v16h b = *(const v16h*)(Wh + (ck1 << 9) + (lane << 4));
            acc1 = __builtin_amdgcn_wmma_f32_16x16x32_f16(
                false, a, false, b, (short)0, acc1, false, false);
        }
    }

    // C/D layout: vgpr v -> M = v + 8*hi, N = lo. Spill partials to LDS as [M][N].
#pragma unroll
    for (int v = 0; v < 8; ++v) {
        const int m = v + 8 * hi;
        part[wave * 256 + m * 16 + lo] = acc0[v] + acc1[v];
    }
    __syncthreads();

    {   // cross-wave reduction + bias; tid -> (m = tid/16, n = tid%16)
        float s = 0.f;
#pragma unroll
        for (int w = 0; w < 8; ++w) s += part[w * 256 + tid];
        const int n = tid & 15;
        if (n < KCLS) s += bias[n];
        logits[tid] = s;
    }
    __syncthreads();

    if (tid < 16) {
        const int pm = tile * 16 + tid;
        if (pm < NCELL) {
            const float* row = &logits[tid * 16];
            float mx = row[0];
#pragma unroll
            for (int n = 1; n < KCLS; ++n) mx = fmaxf(mx, row[n]);
            float ex[KCLS];
            float sum = 0.f;
#pragma unroll
            for (int n = 0; n < KCLS; ++n) { ex[n] = expf(row[n] - mx); sum += ex[n]; }
            const float inv = 1.0f / sum;
#pragma unroll
            for (int n = 0; n < KCLS; ++n) prob_out[pm * KCLS + n] = ex[n] * inv;
        }
    }
}

// ---------------------------------------------------------------------------
// Per-class connected components: 244 synchronous Jacobi min-propagation
// steps entirely in LDS (store mask ? label : BIG so the update is a pure
// 5-point min). One workgroup (32 waves) per class; this is the serial-
// latency tail, so use the WGP-max 1024 threads (~4 cells/thread/step).
// ---------------------------------------------------------------------------
__global__ __launch_bounds__(1024) void cc_kernel(const float* __restrict__ prob,
                                                  float* __restrict__ labels_out,
                                                  int* __restrict__ lab_ws) {
    __shared__ int bufA[NCELL];
    __shared__ int bufB[NCELL];
    const int k   = blockIdx.x;
    const int tid = threadIdx.x;

    for (int m = tid; m < NCELL; m += 1024) {
        const float pv = prob[m * KCLS + k];
        bufA[m] = (pv > THRESH) ? (m + 1) : BIGV;
    }
    __syncthreads();

    int* cur = bufA;
    int* nxt = bufB;
    for (int it = 0; it < 244; ++it) {
        for (int m = tid; m < NCELL; m += 1024) {
            int v = cur[m];
            if (v != BIGV) {
                const int r = m / RGRID;
                const int c = m % RGRID;
                if (r > 0)         v = imin2(v, cur[m - RGRID]);
                if (r < RGRID - 1) v = imin2(v, cur[m + RGRID]);
                if (c > 0)         v = imin2(v, cur[m - 1]);
                if (c < RGRID - 1) v = imin2(v, cur[m + 1]);
            }
            nxt[m] = v;
        }
        __syncthreads();
        int* t = cur; cur = nxt; nxt = t;
    }

    for (int m = tid; m < NCELL; m += 1024) {
        const int v = cur[m];
        const int lab = (v == BIGV) ? 0 : v;
        labels_out[m * KCLS + k] = (float)lab;   // labels shape (R,C,K)
        lab_ws[k * NCELL + m] = lab;
    }
}

// ---------------------------------------------------------------------------
// Segment reductions -> boxes/valid (deterministic integer atomics).
// ---------------------------------------------------------------------------
__global__ void seg_init_kernel(int* rmin, int* cmin, int* rmax, int* cmax, int* cnt) {
    int t = blockIdx.x * blockDim.x + threadIdx.x;
    if (t >= KCLS * NSEG) return;
    rmin[t] = 0x7fffffff; cmin[t] = 0x7fffffff;
    rmax[t] = (int)0x80000000; cmax[t] = (int)0x80000000;
    cnt[t] = 0;
}

__global__ void seg_acc_kernel(const int* __restrict__ lab_ws,
                               int* rmin, int* cmin, int* rmax, int* cmax, int* cnt) {
    int t = blockIdx.x * blockDim.x + threadIdx.x;
    if (t >= KCLS * NCELL) return;
    const int k = t / NCELL;
    const int m = t % NCELL;
    const int seg  = lab_ws[t];
    const int r = m / RGRID;
    const int c = m % RGRID;
    const int base = k * NSEG + seg;
    atomicMin(&rmin[base], r);
    atomicMin(&cmin[base], c);
    atomicMax(&rmax[base], r);
    atomicMax(&cmax[base], c);
    atomicAdd(&cnt[base], 1);
}

__global__ void seg_final_kernel(const int* __restrict__ rmin, const int* __restrict__ cmin,
                                 const int* __restrict__ rmax, const int* __restrict__ cmax,
                                 const int* __restrict__ cnt,
                                 float* __restrict__ boxes_out, float* __restrict__ valid_out) {
    int t = blockIdx.x * blockDim.x + threadIdx.x;
    if (t >= KCLS * NSEG) return;
    const int seg = t % NSEG;
    const bool valid = (cnt[t] > 0) && (seg != 0);
    float* bx = boxes_out + t * 4;
    if (valid) {
        bx[0] = (float)(rmin[t] - 1);
        bx[1] = (float)(cmin[t] - 1);
        bx[2] = (float)(rmax[t] + 1);
        bx[3] = (float)(cmax[t] + 1);
        valid_out[t] = 1.0f;
    } else {
        bx[0] = 0.f; bx[1] = 0.f; bx[2] = 0.f; bx[3] = 0.f;
        valid_out[t] = 0.0f;
    }
}

// ---------------------------------------------------------------------------
extern "C" void kernel_launch(void* const* d_in, const int* in_sizes, int n_in,
                              void* d_out, int out_size, void* d_ws, size_t ws_size,
                              hipStream_t stream) {
    (void)in_sizes; (void)n_in; (void)out_size; (void)ws_size;
    const float* img  = (const float*)d_in[0];   // (1,1024,1024,3)
    const float* W    = (const float*)d_in[1];   // (12288,10)
    const float* bias = (const float*)d_in[2];   // (10,)

    float* out        = (float*)d_out;
    float* prob_out   = out;                       // 37210
    float* labels_out = out + 37210;               // 37210
    float* boxes_out  = out + 74420;               // 148880
    float* valid_out  = out + 223300;              // 37220

    char* ws = (char*)d_ws;
    _Float16* Wh   = (_Float16*)ws;                        // 393216 B
    int*      labw = (int*)(ws + 393216);                  // 148840 B
    size_t off = 393216 + 148864;                          // 64B-aligned
    int* rmin = (int*)(ws + off); off += 148880 + 48;      // each 10*3722 ints
    int* cmin = (int*)(ws + off); off += 148880 + 48;
    int* rmax = (int*)(ws + off); off += 148880 + 48;
    int* cmax = (int*)(ws + off); off += 148880 + 48;
    int* cnt  = (int*)(ws + off);

    // 1) Pack W into f16 B-fragments
    prep_w_kernel<<<(NCHUNK * 512 + 255) / 256, 256, 0, stream>>>(W, Wh);
    // 2) WMMA patch-GEMM + bias + softmax -> prob
    gemm_softmax_kernel<<<NTILE, 256, 0, stream>>>(img, Wh, bias, prob_out);
    // 3) Per-class connected components in LDS -> labels
    cc_kernel<<<KCLS, 1024, 0, stream>>>(prob_out, labels_out, labw);
    // 4) Segment reductions -> boxes / valid
    seg_init_kernel<<<(KCLS * NSEG + 255) / 256, 256, 0, stream>>>(rmin, cmin, rmax, cmax, cnt);
    seg_acc_kernel<<<(KCLS * NCELL + 255) / 256, 256, 0, stream>>>(labw, rmin, cmin, rmax, cmax, cnt);
    seg_final_kernel<<<(KCLS * NSEG + 255) / 256, 256, 0, stream>>>(rmin, cmin, rmax, cmax, cnt,
                                                                    boxes_out, valid_out);
}
